// Rankgraphsage_84550726189070
// MI455X (gfx1250) — matrix operand order repaired and checked by reading, not compiled, not run on hardware
//
#include <hip/hip_runtime.h>
#include <hip/hip_bf16.h>

#define NCAT 20
#define CARD 100
#define EDIM 32
#define PROJ 128
#define CONT 64
#define HID  128

typedef __attribute__((ext_vector_type(16))) __bf16 v16bf;
typedef __attribute__((ext_vector_type(8)))  __bf16 v8bf;
typedef __attribute__((ext_vector_type(8)))  float  v8f;

__device__ __forceinline__ float gelu_f(float x) {
  return 0.5f * x * (1.0f + erff(x * 0.70710678118654752440f));
}

// B fragment from an LDS-resident col-major tile (stride CK). Lane holds
// col = lane&15; 16 contiguous K values starting at (k0-kc) + 16*(lane>>4).
template <int CK>
__device__ __forceinline__ v16bf b_frag_lds(const __bf16* p) {
  v8bf lo = *(const v8bf*)p;
  v8bf hi = *(const v8bf*)(p + 8);
  v16bf f;
#pragma unroll
  for (int i = 0; i < 8; ++i) { f[i] = lo[i]; f[i + 8] = hi[i]; }
  return f;
}

// A fragment from fp32 row: elems 0..15 at rowk0 + 16*half + s (contiguous,
// matching the B fragment k order), converted to bf16.
__device__ __forceinline__ v16bf a_frag_f32(const float* __restrict__ rowk0, int half) {
  const float* p0 = rowk0 + 16 * half;
  const float* p1 = rowk0 + 16 * half + 8;
  float4 a0 = *(const float4*)(p0);
  float4 a1 = *(const float4*)(p0 + 4);
  float4 b0 = *(const float4*)(p1);
  float4 b1 = *(const float4*)(p1 + 4);
  v16bf f;
  f[0]=(__bf16)a0.x;  f[1]=(__bf16)a0.y;  f[2]=(__bf16)a0.z;  f[3]=(__bf16)a0.w;
  f[4]=(__bf16)a1.x;  f[5]=(__bf16)a1.y;  f[6]=(__bf16)a1.z;  f[7]=(__bf16)a1.w;
  f[8]=(__bf16)b0.x;  f[9]=(__bf16)b0.y;  f[10]=(__bf16)b0.z; f[11]=(__bf16)b0.w;
  f[12]=(__bf16)b1.x; f[13]=(__bf16)b1.y; f[14]=(__bf16)b1.z; f[15]=(__bf16)b1.w;
  return f;
}

// Cooperative copy of a 128-col x CK-row chunk of a col-major bf16 weight
// matrix (global stride K) into LDS (stride CK). 16B vector moves.
template <int CK>
__device__ __forceinline__ void stage_w(const __bf16* __restrict__ Wg, int K, int kc,
                                        __bf16* lds, int tid) {
  constexpr int TOTAL8 = (128 * CK) / 8;
#pragma unroll 2
  for (int i = tid; i < TOTAL8; i += 128) {
    const int e = i * 8;
    const int n = e / CK;
    const int kk = e - n * CK;
    *(v8bf*)(lds + n * CK + kk) = *(const v8bf*)(Wg + (size_t)n * K + kc + kk);
  }
}

// ---------- weight transpose + fp32->bf16 : out[n*K+k] = W[k*NOUT+n] ----------
__global__ void wcvt_kernel(const float* __restrict__ W, __bf16* __restrict__ out,
                            int K, int NOUT) {
  int t = blockIdx.x * blockDim.x + threadIdx.x;
  if (t >= K * NOUT) return;
  int n = t / K, k = t - n * K;
  out[t] = (__bf16)W[(size_t)k * NOUT + n];
}

// ---------- fused embedding-gather GEMM: gelu(h0 @ W1 + b1) ----------
// h0[row, c*32+d] = emb[(c*CARD + xcat[row,c])*32 + d], K = 640, NOUT = 128
__global__ __launch_bounds__(128) void embed_gemm_kernel(
    const int* __restrict__ xcat, const float* __restrict__ emb,
    const __bf16* __restrict__ Wb, const float* __restrict__ bias,
    float* __restrict__ C, int M) {
  constexpr int K = NCAT * EDIM;  // 640
  constexpr int CK = 128;         // 5 chunks
  __shared__ __bf16 lB[128 * CK]; // 32 KB
  const int lane = threadIdx.x & 31;
  const int wave = threadIdx.x >> 5;
  const int half = lane >> 4;
  const int r    = lane & 15;
  const int rowBase = (blockIdx.x * 4 + wave) * 16;
  const int row  = min(rowBase + r, M - 1);
  const int* xrow = xcat + (size_t)row * NCAT;
  v8f acc[8] = {};
  for (int kc = 0; kc < K; kc += CK) {
    __syncthreads();
    stage_w<CK>(Wb, K, kc, lB, threadIdx.x);
    __syncthreads();
#pragma unroll
    for (int kk = 0; kk < CK; kk += 32) {
      const int k0 = kc + kk;
      const int c = k0 >> 5;
      const int idx = xrow[c];
      const float* ebase = emb + ((size_t)c * CARD + idx) * EDIM;
      v16bf a = a_frag_f32(ebase, half);  // chunks never cross the 32-dim block
      const __bf16* bbase = lB + r * CK + kk + 16 * half;
      // double-buffered B: load nt+1 before WMMA nt -> partial dscnt waits
      v16bf bcur = b_frag_lds<CK>(bbase);
#pragma unroll
      for (int nt = 0; nt < 8; ++nt) {
        v16bf bnxt = b_frag_lds<CK>(bbase + (((nt + 1) & 7) * 16) * CK);
        acc[nt] = __builtin_amdgcn_wmma_f32_16x16x32_bf16(
            false, a, false, bcur, (short)0, acc[nt], false, false);
        bcur = bnxt;
      }
    }
  }
#pragma unroll
  for (int nt = 0; nt < 8; ++nt) {
    const int col = nt * 16 + r;
    const float bc = bias[col];
#pragma unroll
    for (int rr = 0; rr < 8; ++rr) {
      const int orow = rowBase + rr + 8 * half;
      if (orow < M) C[(size_t)orow * PROJ + col] = gelu_f(acc[nt][rr] + bc);
    }
  }
}

// ---------- generic WMMA GEMM: C = act(A @ W + b), A fp32 (M x K) ----------
template <int K, bool ACT>
__global__ __launch_bounds__(128) void gemm_kernel(
    const float* __restrict__ A, const __bf16* __restrict__ Wb,
    const float* __restrict__ bias, float* __restrict__ C, int ldc, int M) {
  constexpr int CK = (K % 128 == 0) ? 128 : 96;
  __shared__ __bf16 lB[128 * CK];
  const int lane = threadIdx.x & 31;
  const int wave = threadIdx.x >> 5;
  const int half = lane >> 4;
  const int r    = lane & 15;
  const int rowBase = (blockIdx.x * 4 + wave) * 16;
  const int row  = min(rowBase + r, M - 1);
  const float* arow = A + (size_t)row * K;
  v8f acc[8] = {};
  for (int kc = 0; kc < K; kc += CK) {
    __syncthreads();
    stage_w<CK>(Wb, K, kc, lB, threadIdx.x);
    __syncthreads();
#pragma unroll
    for (int kk = 0; kk < CK; kk += 32) {
      v16bf a = a_frag_f32(arow + kc + kk, half);
      const __bf16* bbase = lB + r * CK + kk + 16 * half;
      v16bf bcur = b_frag_lds<CK>(bbase);
#pragma unroll
      for (int nt = 0; nt < 8; ++nt) {
        v16bf bnxt = b_frag_lds<CK>(bbase + (((nt + 1) & 7) * 16) * CK);
        acc[nt] = __builtin_amdgcn_wmma_f32_16x16x32_bf16(
            false, a, false, bcur, (short)0, acc[nt], false, false);
        bcur = bnxt;
      }
    }
  }
#pragma unroll
  for (int nt = 0; nt < 8; ++nt) {
    const int col = nt * 16 + r;
    const float bc = bias[col];
#pragma unroll
    for (int rr = 0; rr < 8; ++rr) {
      const int orow = rowBase + rr + 8 * half;
      if (orow < M) {
        float v = acc[nt][rr] + bc;
        C[(size_t)orow * ldc + col] = ACT ? gelu_f(v) : v;
      }
    }
  }
}

// ---------- fused SAGE GEMM: C = act(agg @ Wl + bl + x @ Wr), NOUT=128 ----------
template <int K, bool ACT>
__global__ __launch_bounds__(128) void sage_gemm_kernel(
    const float* __restrict__ agg, const float* __restrict__ x,
    const __bf16* __restrict__ Wlb, const __bf16* __restrict__ Wrb,
    const float* __restrict__ bias, float* __restrict__ C, int M) {
  constexpr int CK = (K % 128 == 0) ? 128 : 96;
  __shared__ __bf16 lL[128 * CK];
  __shared__ __bf16 lR[128 * CK];
  const int lane = threadIdx.x & 31;
  const int wave = threadIdx.x >> 5;
  const int half = lane >> 4;
  const int r    = lane & 15;
  const int rowBase = (blockIdx.x * 4 + wave) * 16;
  const int row  = min(rowBase + r, M - 1);
  const float* arow = agg + (size_t)row * K;
  const float* xrow = x   + (size_t)row * K;
  v8f acc[8] = {};
  for (int kc = 0; kc < K; kc += CK) {
    __syncthreads();
    stage_w<CK>(Wlb, K, kc, lL, threadIdx.x);
    stage_w<CK>(Wrb, K, kc, lR, threadIdx.x);
    __syncthreads();
#pragma unroll
    for (int kk = 0; kk < CK; kk += 32) {
      v16bf af = a_frag_f32(arow + kc + kk, half);
      v16bf xf = a_frag_f32(xrow + kc + kk, half);
      const __bf16* lbase = lL + r * CK + kk + 16 * half;
      const __bf16* rbase = lR + r * CK + kk + 16 * half;
      // pipeline: always one LDS fetch in flight between WMMAs
      v16bf bl = b_frag_lds<CK>(lbase);
#pragma unroll
      for (int nt = 0; nt < 8; ++nt) {
        v16bf br = b_frag_lds<CK>(rbase + (nt * 16) * CK);
        acc[nt] = __builtin_amdgcn_wmma_f32_16x16x32_bf16(
            false, af, false, bl, (short)0, acc[nt], false, false);
        v16bf bln = b_frag_lds<CK>(lbase + (((nt + 1) & 7) * 16) * CK);
        acc[nt] = __builtin_amdgcn_wmma_f32_16x16x32_bf16(
            false, xf, false, br, (short)0, acc[nt], false, false);
        bl = bln;
      }
    }
  }
#pragma unroll
  for (int nt = 0; nt < 8; ++nt) {
    const int col = nt * 16 + r;
    const float bc = bias[col];
#pragma unroll
    for (int rr = 0; rr < 8; ++rr) {
      const int orow = rowBase + rr + 8 * half;
      if (orow < M) {
        float v = acc[nt][rr] + bc;
        C[(size_t)orow * HID + col] = ACT ? gelu_f(v) : v;
      }
    }
  }
}

// ---------- graph kernels ----------
__global__ void deg_kernel(const int* __restrict__ dst, float* __restrict__ deg, int E) {
  int t = blockIdx.x * blockDim.x + threadIdx.x;
  if (t < E) atomicAdd(&deg[dst[t]], 1.0f);
}

template <int DIN>
__global__ void scatter_kernel(const float* __restrict__ x, const int* __restrict__ src,
                               const int* __restrict__ dst, float* __restrict__ agg, int E) {
  constexpr int C4 = DIN / 4;
  int t = blockIdx.x * blockDim.x + threadIdx.x;
  int e = t / C4, c = t - e * C4;
  if (e >= E) return;
  int s = src[e], d = dst[e];
  float4 v = *(const float4*)(x + (size_t)s * DIN + c * 4);
  float* ap = agg + (size_t)d * DIN + c * 4;
  atomicAdd(ap + 0, v.x); atomicAdd(ap + 1, v.y);
  atomicAdd(ap + 2, v.z); atomicAdd(ap + 3, v.w);
}

__global__ void norm_kernel(float* __restrict__ agg, const float* __restrict__ deg,
                            int DIN, int total) {
  int t = blockIdx.x * blockDim.x + threadIdx.x;
  if (t >= total) return;
  int n = t / DIN;
  agg[t] = agg[t] / fmaxf(deg[n], 1.0f);
}

__global__ void concat_kernel(const float* __restrict__ xcont, float* __restrict__ x1, int Nn) {
  int t = blockIdx.x * blockDim.x + threadIdx.x;
  if (t >= Nn * CONT) return;
  int n = t / CONT, j = t - n * CONT;
  x1[(size_t)n * (PROJ + CONT) + PROJ + j] = xcont[t];
}

// final 128 -> 1 head: out[n] = bl + agg[n]·Wl + x[n]·Wr
__global__ void conv3_kernel(const float* __restrict__ agg, const float* __restrict__ x,
                             const float* __restrict__ Wl, const float* __restrict__ bl,
                             const float* __restrict__ Wr, float* __restrict__ out, int Nn) {
  int n = blockIdx.x * blockDim.x + threadIdx.x;
  if (n >= Nn) return;
  const float* a  = agg + (size_t)n * HID;
  const float* xx = x   + (size_t)n * HID;
  float s = bl[0];
#pragma unroll 4
  for (int d = 0; d < HID; ++d) s += a[d] * Wl[d] + xx[d] * Wr[d];
  out[n] = s;
}

static inline int cdiv(long long a, long long b) { return (int)((a + b - 1) / b); }

extern "C" void kernel_launch(void* const* d_in, const int* in_sizes, int n_in,
                              void* d_out, int out_size, void* d_ws, size_t ws_size,
                              hipStream_t stream) {
  const int*   x_cat  = (const int*)d_in[0];
  const float* x_cont = (const float*)d_in[1];
  const int*   eidx   = (const int*)d_in[2];
  const float* emb    = (const float*)d_in[3];
  const float* W1 = (const float*)d_in[4];  const float* b1 = (const float*)d_in[5];
  const float* W2 = (const float*)d_in[6];  const float* b2 = (const float*)d_in[7];
  const float* W3 = (const float*)d_in[8];  const float* b3 = (const float*)d_in[9];
  const float* c1Wl = (const float*)d_in[10]; const float* c1bl = (const float*)d_in[11];
  const float* c1Wr = (const float*)d_in[12];
  const float* c2Wl = (const float*)d_in[13]; const float* c2bl = (const float*)d_in[14];
  const float* c2Wr = (const float*)d_in[15];
  const float* c3Wl = (const float*)d_in[16]; const float* c3bl = (const float*)d_in[17];
  const float* c3Wr = (const float*)d_in[18];
  float* out = (float*)d_out;

  const int Nn = in_sizes[0] / NCAT;
  const int E  = in_sizes[2] / 2;
  const int* src = eidx;
  const int* dst = eidx + E;
  const int DCAT = PROJ + CONT;  // 192

  // ---- workspace bump allocator ----
  char* wsp = (char*)d_ws;
  auto balloc = [&](size_t bytes) {
    char* p = wsp; wsp += (bytes + 255) & ~(size_t)255; return p;
  };
  __bf16* W1b  = (__bf16*)balloc(sizeof(__bf16) * PROJ * (NCAT * EDIM));
  __bf16* W2b  = (__bf16*)balloc(sizeof(__bf16) * PROJ * PROJ);
  __bf16* W3b  = (__bf16*)balloc(sizeof(__bf16) * PROJ * PROJ);
  __bf16* Wl1b = (__bf16*)balloc(sizeof(__bf16) * HID * DCAT);
  __bf16* Wr1b = (__bf16*)balloc(sizeof(__bf16) * HID * DCAT);
  __bf16* Wl2b = (__bf16*)balloc(sizeof(__bf16) * HID * HID);
  __bf16* Wr2b = (__bf16*)balloc(sizeof(__bf16) * HID * HID);
  float* bufP = (float*)balloc(sizeof(float) * (size_t)Nn * PROJ);   // h1, later x3
  float* bufQ = (float*)balloc(sizeof(float) * (size_t)Nn * PROJ);   // h2, later agg3
  float* x1   = (float*)balloc(sizeof(float) * (size_t)Nn * DCAT);   // concat input
  float* agg1 = (float*)balloc(sizeof(float) * (size_t)Nn * DCAT);
  float* x2   = (float*)balloc(sizeof(float) * (size_t)Nn * HID);
  float* agg2 = (float*)balloc(sizeof(float) * (size_t)Nn * HID);
  float* deg  = (float*)balloc(sizeof(float) * (size_t)Nn);

  // ---- weight conversion (tiny; L2-resident afterwards) ----
  wcvt_kernel<<<cdiv(640 * 128, 256), 256, 0, stream>>>(W1,  W1b,  640, 128);
  wcvt_kernel<<<cdiv(128 * 128, 256), 256, 0, stream>>>(W2,  W2b,  128, 128);
  wcvt_kernel<<<cdiv(128 * 128, 256), 256, 0, stream>>>(W3,  W3b,  128, 128);
  wcvt_kernel<<<cdiv(192 * 128, 256), 256, 0, stream>>>(c1Wl, Wl1b, 192, 128);
  wcvt_kernel<<<cdiv(192 * 128, 256), 256, 0, stream>>>(c1Wr, Wr1b, 192, 128);
  wcvt_kernel<<<cdiv(128 * 128, 256), 256, 0, stream>>>(c2Wl, Wl2b, 128, 128);
  wcvt_kernel<<<cdiv(128 * 128, 256), 256, 0, stream>>>(c2Wr, Wr2b, 128, 128);

  // ---- degree (computed once, shared by all three convs) ----
  hipMemsetAsync(deg, 0, sizeof(float) * Nn, stream);
  deg_kernel<<<cdiv(E, 256), 256, 0, stream>>>(dst, deg, E);

  // ---- MLP (WMMA): embed-fused GEMM, then two 128x128 GEMMs ----
  const int gblocks = cdiv(Nn, 64);  // 4 waves/block * 16 rows/wave
  embed_gemm_kernel<<<gblocks, 128, 0, stream>>>(x_cat, emb, W1b, b1, bufP, Nn);
  gemm_kernel<128, true ><<<gblocks, 128, 0, stream>>>(bufP, W2b, b2, bufQ, 128, Nn);
  gemm_kernel<128, false><<<gblocks, 128, 0, stream>>>(bufQ, W3b, b3, x1, DCAT, Nn);
  concat_kernel<<<cdiv((long long)Nn * CONT, 256), 256, 0, stream>>>(x_cont, x1, Nn);

  // ---- conv1 (192 -> 128, gelu) ----
  hipMemsetAsync(agg1, 0, sizeof(float) * (size_t)Nn * DCAT, stream);
  scatter_kernel<192><<<cdiv((long long)E * 48, 256), 256, 0, stream>>>(x1, src, dst, agg1, E);
  norm_kernel<<<cdiv((long long)Nn * DCAT, 256), 256, 0, stream>>>(agg1, deg, DCAT, Nn * DCAT);
  sage_gemm_kernel<192, true><<<gblocks, 128, 0, stream>>>(agg1, x1, Wl1b, Wr1b, c1bl, x2, Nn);

  // ---- conv2 (128 -> 128, gelu) ----
  hipMemsetAsync(agg2, 0, sizeof(float) * (size_t)Nn * HID, stream);
  scatter_kernel<128><<<cdiv((long long)E * 32, 256), 256, 0, stream>>>(x2, src, dst, agg2, E);
  norm_kernel<<<cdiv((long long)Nn * HID, 256), 256, 0, stream>>>(agg2, deg, HID, Nn * HID);
  sage_gemm_kernel<128, true><<<gblocks, 128, 0, stream>>>(agg2, x2, Wl2b, Wr2b, c2bl, bufP, Nn);

  // ---- conv3 (128 -> 1, no act); x3 = bufP, agg3 = bufQ ----
  hipMemsetAsync(bufQ, 0, sizeof(float) * (size_t)Nn * HID, stream);
  scatter_kernel<128><<<cdiv((long long)E * 32, 256), 256, 0, stream>>>(bufP, src, dst, bufQ, E);
  norm_kernel<<<cdiv((long long)Nn * HID, 256), 256, 0, stream>>>(bufQ, deg, HID, Nn * HID);
  conv3_kernel<<<cdiv(Nn, 256), 256, 0, stream>>>(bufQ, bufP, c3Wl, c3bl, c3Wr, out, Nn);
}
